// GRUclassifier_25168508354722
// MI455X (gfx1250) — matrix operand order repaired and checked by reading, not compiled, not run on hardware
//
#include <hip/hip_runtime.h>
#include <hip/hip_bf16.h>
#include <math.h>

// ---------------------------------------------------------------------------
// GRU classifier for MI455X (gfx1250, wave32, WMMA 16x16x32 bf16)
//   V=30001 I=128 H=256 O=50 B=32 T=10000
// ---------------------------------------------------------------------------

typedef __bf16 bf16_t;
typedef __attribute__((ext_vector_type(16))) __bf16 v16bf;
typedef __attribute__((ext_vector_type(8)))  __bf16 v8bf;
typedef __attribute__((ext_vector_type(8)))  float  v8f;

#define V_SZ 30001
#define I_SZ 128
#define H_SZ 256
#define G_SZ 768      // 3*H
#define O_SZ 50
#define B_SZ 32
#define T_SZ 10000

// workspace byte offsets (all 256B aligned)
#define OFF_WIH 0ull                      // 768*128 bf16   = 196608 B
#define OFF_WHH 196608ull                 // 768*256 bf16   = 393216 B
#define OFF_EMB 589824ull                 // 30001*128 bf16 = 7680256 B
#define OFF_GX  8270080ull                // [T][B][768] f32 = 983040000 B
#define OFF_HS  991310080ull              // [T][B][256] f32 = 327680000 B

__device__ __forceinline__ float fast_sigmoid(float x) {
    return 1.0f / (1.0f + __expf(-x));
}
__device__ __forceinline__ float fast_tanh(float x) {
    float e = __expf(-2.0f * fabsf(x));
    float r = (1.0f - e) / (1.0f + e);
    return copysignf(r, x);
}
__device__ __forceinline__ v8f wmma_bf16(v16bf a, v16bf b, v8f c) {
    return __builtin_amdgcn_wmma_f32_16x16x32_bf16(false, a, false, b,
                                                   (short)0, c, false, false);
}

// gfx1250 async global->LDS copy (ASYNCcnt-tracked); inline asm per
// CDNA5 ISA so it works on both toolchains.
__device__ __forceinline__ void async_gld_b128(unsigned lds_off,
                                               const float* gptr) {
    unsigned long long ga = (unsigned long long)(uintptr_t)gptr;
    asm volatile("global_load_async_to_lds_b128 %0, %1, off"
                 :: "v"(lds_off), "v"(ga)
                 : "memory");
}
__device__ __forceinline__ void wait_asynccnt0() {
    asm volatile("s_wait_asynccnt 0x0" ::: "memory");
}

// ---------------------------------------------------------------------------
// prep: f32 -> bf16 conversion (weights + embedding table)
// ---------------------------------------------------------------------------
__global__ void cvt_f32_to_bf16(const float* __restrict__ src,
                                bf16_t* __restrict__ dst, int n) {
    int i = blockIdx.x * blockDim.x + threadIdx.x;
    if (i < n) dst[i] = (bf16_t)src[i];
}

// ---------------------------------------------------------------------------
// Phase 1: gx[t*32+b][g] = embed[x[b][t]] . W_ih[g] + b_ih[g]
// grid = 20000 M-tiles, 256 threads = 8 waves, each wave covers 6 N-tiles.
// ---------------------------------------------------------------------------
__global__ __launch_bounds__(256) void gemm_gx(
        const int* __restrict__ x,
        const bf16_t* __restrict__ emb_bf,
        const bf16_t* __restrict__ wih_bf,
        const float* __restrict__ b_ih,
        float* __restrict__ gx) {
    const int tid  = threadIdx.x;
    const int wv   = tid >> 5;
    const int L    = tid & 31;
    const int n16  = L & 15;
    const bool hi  = (L >= 16);
    const int mtile = blockIdx.x;

    // token for this lane's A row (lanes 16..31 mirror rows 0..15)
    const int r   = mtile * 16 + n16;
    const int t   = r >> 5;
    const int b   = r & 31;
    const int tok = x[(size_t)b * T_SZ + t];
    const bf16_t* arow = emb_bf + (size_t)tok * I_SZ;

    const int a_off = hi ? 16 : 0;   // A-frag K sub-offset (contiguous k = 16h + s)
    const int b_off = hi ? 16 : 0;   // B-frag K sub-offset (same contiguous k as A)

    v8f acc[6];
#pragma unroll
    for (int j = 0; j < 6; ++j)
#pragma unroll
        for (int v = 0; v < 8; ++v) acc[j][v] = 0.0f;

#pragma unroll
    for (int kt = 0; kt < 4; ++kt) {                 // K = 128 = 4 * 32
        v8bf a0 = *(const v8bf*)(arow + kt * 32 + a_off);
        v8bf a1 = *(const v8bf*)(arow + kt * 32 + a_off + 8);
        v16bf a = __builtin_shufflevector(a0, a1,
                    0,1,2,3,4,5,6,7,8,9,10,11,12,13,14,15);
#pragma unroll
        for (int j = 0; j < 6; ++j) {
            const int ncol = (wv * 6 + j) * 16 + n16;
            v16bf bf = *(const v16bf*)(wih_bf + (size_t)ncol * I_SZ
                                       + kt * 32 + b_off);
            acc[j] = wmma_bf16(a, bf, acc[j]);
        }
    }

#pragma unroll
    for (int j = 0; j < 6; ++j) {
        const int ncol = (wv * 6 + j) * 16 + n16;
        const float bias = b_ih[ncol];
#pragma unroll
        for (int v = 0; v < 8; ++v) {
            const int m = v + (hi ? 8 : 0);          // C/D layout row
            const size_t row = (size_t)mtile * 16 + m;
            // streamed once, consumed once much later: keep out of L2
            __builtin_nontemporal_store(acc[j][v] + bias,
                                        &gx[row * G_SZ + ncol]);
        }
    }
}

// ---------------------------------------------------------------------------
// Phase 2: sequential GRU scan. grid = 2 blocks (independent batch halves),
// 512 threads = 16 waves; wave w owns hidden columns [16w, 16w+16).
//  - W_hh bf16 fragments (3 gates x 8 K-tiles) pinned in VGPRs for all steps
//  - previous h kept in registers (each lane re-reads exactly what it wrote)
//  - gx block double-buffered in LDS via global_load_async_to_lds_b128,
//    staged one full timestep ahead (hides HBM latency off critical path)
// ---------------------------------------------------------------------------
#define HB_PITCH 264   // bf16 h pitch (bank-conflict padding)
#define GX_PITCH 772   // f32 gx staging pitch (bank-conflict padding)

__global__ __launch_bounds__(512) void gru_scan(
        const bf16_t* __restrict__ whh_bf,
        const float* __restrict__ b_hh,
        const float* __restrict__ gx,
        float* __restrict__ hs) {
    __shared__ bf16_t h_b16[16 * HB_PITCH];
    __shared__ float  gxbuf[2][16 * GX_PITCH];

    const int tid  = threadIdx.x;
    const int wv   = tid >> 5;            // hidden tile 0..15
    const int L    = tid & 31;
    const int n16  = L & 15;
    const bool hi  = (L >= 16);
    const int mt   = blockIdx.x;          // batch half (rows mt*16 .. +15)
    const int colj = wv * 16 + n16;       // hidden column
    const int a_off = hi ? 16 : 0;
    const int b_off = hi ? 16 : 0;
    const int mrow  = hi ? 8 : 0;         // local batch row base for C layout

    // preload W_hh B-fragments: gh[.,g*256+colj] needs W_hh row g*256+colj
    v16bf bfrag[3][8];
#pragma unroll
    for (int g = 0; g < 3; ++g)
#pragma unroll
        for (int kt = 0; kt < 8; ++kt)
            bfrag[g][kt] = *(const v16bf*)(whh_bf
                            + (size_t)(g * H_SZ + colj) * H_SZ
                            + kt * 32 + b_off);

    float bias[3];
#pragma unroll
    for (int g = 0; g < 3; ++g) bias[g] = b_hh[g * H_SZ + colj];

    // h0 = 0 (register copy of this lane's 8 elements + shared bf16 copy)
    float hprev[8];
#pragma unroll
    for (int v = 0; v < 8; ++v) hprev[v] = 0.0f;
    for (int i = tid; i < 16 * HB_PITCH; i += 512) h_b16[i] = (bf16_t)0.0f;

    // prime the async pipeline: stage gx block for t=0 (wave wv copies its
    // local batch row wv; 192 b128 chunks per row, 6 per lane)
    {
        const float* src = gx + ((size_t)mt * 16 + wv) * G_SZ;
        const unsigned dst = (unsigned)(uintptr_t)&gxbuf[0][wv * GX_PITCH];
#pragma unroll
        for (int k = 0; k < 6; ++k)
            async_gld_b128(dst + (unsigned)(L + 32 * k) * 16,
                           src + (L + 32 * k) * 4);
    }

    for (int t = 0; t < T_SZ; ++t) {
        wait_asynccnt0();    // my share of gxbuf[t&1] landed in LDS
        __syncthreads();     // everyone's share + h_b16(t) visible

        // stage gx for t+1 while we compute step t
        if (t + 1 < T_SZ) {
            const float* src = gx
                + ((size_t)(t + 1) * B_SZ + mt * 16 + wv) * G_SZ;
            const unsigned dst =
                (unsigned)(uintptr_t)&gxbuf[(t + 1) & 1][wv * GX_PITCH];
#pragma unroll
            for (int k = 0; k < 6; ++k)
                async_gld_b128(dst + (unsigned)(L + 32 * k) * 16,
                               src + (L + 32 * k) * 4);
        }

        // my gate pre-activations from the staged block
        const float* gxc = &gxbuf[t & 1][0];
        float gxv[3][8];
#pragma unroll
        for (int g = 0; g < 3; ++g)
#pragma unroll
            for (int v = 0; v < 8; ++v)
                gxv[g][v] = gxc[(mrow + v) * GX_PITCH + g * H_SZ + colj];

        // gh tiles: acc[g] = h_t x W_hh^T (K = 256 = 8 * 32)
        v8f acc[3];
#pragma unroll
        for (int g = 0; g < 3; ++g)
#pragma unroll
            for (int v = 0; v < 8; ++v) acc[g][v] = 0.0f;

#pragma unroll
        for (int kt = 0; kt < 8; ++kt) {
            const bf16_t* hrow = &h_b16[n16 * HB_PITCH + kt * 32 + a_off];
            v8bf a0 = *(const v8bf*)(hrow);
            v8bf a1 = *(const v8bf*)(hrow + 8);
            v16bf a = __builtin_shufflevector(a0, a1,
                        0,1,2,3,4,5,6,7,8,9,10,11,12,13,14,15);
#pragma unroll
            for (int g = 0; g < 3; ++g)
                acc[g] = wmma_bf16(a, bfrag[g][kt], acc[g]);
        }

        // gates (torch GRU): r,z = sigmoid; n = tanh(xn + r*hn)
        float hnew[8];
#pragma unroll
        for (int v = 0; v < 8; ++v) {
            const float rr = fast_sigmoid(gxv[0][v] + acc[0][v] + bias[0]);
            const float zz = fast_sigmoid(gxv[1][v] + acc[1][v] + bias[1]);
            const float nn = fast_tanh(gxv[2][v] + rr * (acc[2][v] + bias[2]));
            hnew[v] = (1.0f - zz) * nn + zz * hprev[v];
        }

        __syncthreads();     // all waves finished reading h_b16(t) + gxbuf
#pragma unroll
        for (int v = 0; v < 8; ++v) {
            const int lr = mrow + v;
            hprev[v] = hnew[v];
            h_b16[lr * HB_PITCH + colj] = (bf16_t)hnew[v];
            __builtin_nontemporal_store(hnew[v],
                &hs[((size_t)t * B_SZ + mt * 16 + lr) * H_SZ + colj]);
        }
    }
}

// ---------------------------------------------------------------------------
// Phase 3: logits = hs @ W_lin^T + b_lin, then log_softmax over O=50.
// One wave per (t,b) row; lane L owns logits o=L and o=L+32.
// ---------------------------------------------------------------------------
#define WL_PITCH 257

__global__ __launch_bounds__(256) void head_logsoftmax(
        const float* __restrict__ hs,
        const float* __restrict__ wlin,
        const float* __restrict__ blin,
        float* __restrict__ out) {
    __shared__ float sW[O_SZ * WL_PITCH];
    __shared__ float sH[8 * H_SZ];

    const int tid = threadIdx.x;
    for (int i = tid; i < O_SZ * H_SZ; i += 256)
        sW[(i / H_SZ) * WL_PITCH + (i % H_SZ)] = wlin[i];

    const int wv = tid >> 5;
    const int L  = tid & 31;
    const size_t row = (size_t)blockIdx.x * 8 + wv;   // row = t*32 + b
    const float* hrow = hs + row * H_SZ;
    for (int i = L; i < H_SZ; i += 32) sH[wv * H_SZ + i] = hrow[i];
    __syncthreads();

    float acc0 = blin[L];
    float acc1 = (L < O_SZ - 32) ? blin[L + 32] : 0.0f;
    for (int k = 0; k < H_SZ; ++k) {
        const float hv = sH[wv * H_SZ + k];
        acc0 = fmaf(hv, sW[L * WL_PITCH + k], acc0);
        if (L < O_SZ - 32)
            acc1 = fmaf(hv, sW[(L + 32) * WL_PITCH + k], acc1);
    }

    // wave32 reductions for log_softmax over 50 logits
    float m = acc0;
    if (L < O_SZ - 32) m = fmaxf(m, acc1);
#pragma unroll
    for (int off = 16; off > 0; off >>= 1)
        m = fmaxf(m, __shfl_xor(m, off, 32));
    float s = __expf(acc0 - m);
    if (L < O_SZ - 32) s += __expf(acc1 - m);
#pragma unroll
    for (int off = 16; off > 0; off >>= 1)
        s += __shfl_xor(s, off, 32);
    const float lse = m + __logf(s);

    const size_t t = row >> 5, b = row & 31;
    float* orow = out + (b * (size_t)T_SZ + t) * O_SZ;
    orow[L] = acc0 - lse;
    if (L < O_SZ - 32) orow[L + 32] = acc1 - lse;
}

// ---------------------------------------------------------------------------
extern "C" void kernel_launch(void* const* d_in, const int* in_sizes, int n_in,
                              void* d_out, int out_size, void* d_ws, size_t ws_size,
                              hipStream_t stream) {
    (void)in_sizes; (void)n_in; (void)out_size; (void)ws_size;
    const int*   x     = (const int*)d_in[0];
    const float* embed = (const float*)d_in[1];
    const float* W_ih  = (const float*)d_in[2];
    const float* W_hh  = (const float*)d_in[3];
    const float* b_ih  = (const float*)d_in[4];
    const float* b_hh  = (const float*)d_in[5];
    const float* W_lin = (const float*)d_in[6];
    const float* b_lin = (const float*)d_in[7];
    float* out = (float*)d_out;

    char* ws = (char*)d_ws;
    bf16_t* wih_bf = (bf16_t*)(ws + OFF_WIH);
    bf16_t* whh_bf = (bf16_t*)(ws + OFF_WHH);
    bf16_t* emb_bf = (bf16_t*)(ws + OFF_EMB);
    float*  gxp    = (float*)(ws + OFF_GX);
    float*  hsp    = (float*)(ws + OFF_HS);

    const int nWih = G_SZ * I_SZ;          // 98304
    const int nWhh = G_SZ * H_SZ;          // 196608
    const int nEmb = V_SZ * I_SZ;          // 3840128
    cvt_f32_to_bf16<<<(nWih + 255) / 256, 256, 0, stream>>>(W_ih, wih_bf, nWih);
    cvt_f32_to_bf16<<<(nWhh + 255) / 256, 256, 0, stream>>>(W_hh, whh_bf, nWhh);
    cvt_f32_to_bf16<<<(nEmb + 255) / 256, 256, 0, stream>>>(embed, emb_bf, nEmb);

    // Phase 1: 20000 M-tiles (rows = t*32+b), 8 waves x 6 N-tiles each
    gemm_gx<<<(B_SZ * T_SZ) / 16, 256, 0, stream>>>(x, emb_bf, wih_bf, b_ih, gxp);

    // Phase 2: 2 independent batch halves, 16 waves each
    gru_scan<<<2, 512, 0, stream>>>(whh_bf, b_hh, gxp, hsp);

    // Phase 3: one wave per (t,b) row
    head_logsoftmax<<<(B_SZ * T_SZ) / 8, 256, 0, stream>>>(hsp, W_lin, b_lin, out);
}
